// SelfAttention_8280696947408
// MI455X (gfx1250) — matrix operand order repaired and checked by reading, not compiled, not run on hardware
//
#include <hip/hip_runtime.h>
#include <hip/hip_bf16.h>

typedef __attribute__((ext_vector_type(2)))  float  v2f;
typedef __attribute__((ext_vector_type(8)))  float  v8f;
typedef __attribute__((ext_vector_type(8)))  __bf16 v8bf;
typedef __attribute__((ext_vector_type(16))) __bf16 v16bf;

#define B_   4
#define N_   4096
#define C_   64
#define CF_  8

// ---------------------------------------------------------------------------
// Kernel 1: per-pixel linear projections.
//   f = x@w_f + b_f  -> fA  [B*N][8]  f32   (scores A-operand source)
//   g = x@w_g + b_g  -> gB  [B*N][8]  f32   (scores B-operand source)
//   h = x@w_h + b_h  -> hT  [B][C][N] bf16  (transposed: A-operand of P@h gemm
//                                            becomes contiguous 16B loads)
// One block per pixel row (64 threads), x row staged in LDS.
// ---------------------------------------------------------------------------
__global__ __launch_bounds__(64)
void proj_kernel(const float* __restrict__ x,
                 const float* __restrict__ wf, const float* __restrict__ bf,
                 const float* __restrict__ wg, const float* __restrict__ bg,
                 const float* __restrict__ wh, const float* __restrict__ bh,
                 float* __restrict__ fA, float* __restrict__ gB,
                 __bf16* __restrict__ hT)
{
    __shared__ float xr[C_];
    const int row = blockIdx.x;          // 0 .. B*N-1
    const int t   = threadIdx.x;         // 0 .. 63
    xr[t] = x[(size_t)row * C_ + t];
    __syncthreads();

    // h[row][t]
    float acc = bh[t];
#pragma unroll
    for (int i = 0; i < C_; ++i)
        acc = fmaf(xr[i], wh[i * C_ + t], acc);
    const int b = row >> 12;             // / 4096
    const int n = row & (N_ - 1);
    hT[((size_t)b * C_ + t) * N_ + n] = (__bf16)acc;

    if (t < CF_) {
        float a = bf[t];
#pragma unroll
        for (int i = 0; i < C_; ++i)
            a = fmaf(xr[i], wf[i * CF_ + t], a);
        fA[(size_t)row * CF_ + t] = a;
    } else if (t < 2 * CF_) {
        const int c = t - CF_;
        float a = bg[c];
#pragma unroll
        for (int i = 0; i < C_; ++i)
            a = fmaf(xr[i], wg[i * CF_ + c], a);
        gB[(size_t)row * CF_ + c] = a;
    }
}

// ---------------------------------------------------------------------------
// Kernel 2: flash attention. One wave32 owns 16 queries of one batch.
//
// Per 32-key tile:
//   S^T (keys x queries): 2x [v_wmma_f32_16x16x4_f32 x2, K=8]   (f32 logits)
//   online softmax (running max/sum; lane<->lane+16 pairing via shfl_xor 16)
//   out^T (ch x queries): 4x v_wmma_f32_16x16x32_bf16, A = h^T tile (16B
//   contiguous bf16 loads), B = probabilities (assembled in-register).
// ---------------------------------------------------------------------------
__global__ __launch_bounds__(256)
void attn_kernel(const float* __restrict__ x,
                 const float* __restrict__ fA,
                 const float* __restrict__ gB,
                 const __bf16* __restrict__ hT,
                 const float* __restrict__ gammaPtr,
                 float* __restrict__ out)
{
    const int lane = threadIdx.x & 31;
    const int wid  = threadIdx.x >> 5;
    const int tile = blockIdx.x * 8 + wid;      // 0 .. 1023
    const int b    = tile >> 8;                 // / (N/16)
    const int qt   = tile & 255;                // query tile within batch
    const int l16  = lane & 15;
    const bool hiH = lane >= 16;
    const int cb0  = hiH ? 2 : 0;               // f32 WMMA K split per lane-half

    // g tile -> B operand of the scores gemm (fixed for the whole wave).
    // B 4x16 f32 layout: vgpr0 = K0(l<16)/K2(l>=16), vgpr1 = K1/K3.
    const float* gRow = gB + ((size_t)b * N_ + (size_t)qt * 16 + l16) * CF_;
    const v2f bg0 = { gRow[cb0],     gRow[cb0 + 1] };     // K 0..3
    const v2f bg1 = { gRow[4 + cb0], gRow[5 + cb0] };     // K 4..7

    v8f acc[4];
#pragma unroll
    for (int ct = 0; ct < 4; ++ct)
#pragma unroll
        for (int r = 0; r < 8; ++r) acc[ct][r] = 0.0f;

    float mRun = -__builtin_inff();
    float lRun = 0.0f;

    const float*  fBase = fA + (size_t)b * N_ * CF_;
    const __bf16* hTb   = hT + (size_t)b * C_ * N_;

    for (int kb = 0; kb < N_ / 32; ++kb) {
        // ---- f tile -> A operands (16x4 f32 layout, two K-halves x 2 chunks)
        const float* fR0 = fBase + ((size_t)kb * 32 + l16) * CF_;
        const float* fR1 = fR0 + 16 * CF_;
        __builtin_prefetch(fR0 + 32 * CF_, 0, 0);         // next key tile
        const v2f a00 = { fR0[cb0],     fR0[cb0 + 1] };
        const v2f a01 = { fR0[4 + cb0], fR0[5 + cb0] };
        const v2f a10 = { fR1[cb0],     fR1[cb0 + 1] };
        const v2f a11 = { fR1[4 + cb0], fR1[5 + cb0] };

        // ---- S^T tiles: s0 = keys kb*32+0..15, s1 = keys kb*32+16..31
        v8f s0, s1;
#pragma unroll
        for (int r = 0; r < 8; ++r) { s0[r] = 0.0f; s1[r] = 0.0f; }
        s0 = __builtin_amdgcn_wmma_f32_16x16x4_f32(false, a00, false, bg0,
                                                   (short)0, s0, false, false);
        s0 = __builtin_amdgcn_wmma_f32_16x16x4_f32(false, a01, false, bg1,
                                                   (short)0, s0, false, false);
        s1 = __builtin_amdgcn_wmma_f32_16x16x4_f32(false, a10, false, bg0,
                                                   (short)0, s1, false, false);
        s1 = __builtin_amdgcn_wmma_f32_16x16x4_f32(false, a11, false, bg1,
                                                   (short)0, s1, false, false);

        // ---- online softmax. Lane pair (l, l+16) shares one query; together
        // they hold all 32 keys of this tile for that query.
        float tmax = -__builtin_inff();
#pragma unroll
        for (int r = 0; r < 8; ++r) {
            tmax = fmaxf(tmax, s0[r]);
            tmax = fmaxf(tmax, s1[r]);
        }
        tmax = fmaxf(tmax, __shfl_xor(tmax, 16, 32));
        const float mNew = fmaxf(mRun, tmax);
        const float corr = __expf(mRun - mNew);   // exp(-inf)=0 on first tile
        mRun = mNew;

        float e0[8], e1[8];
        float ls = 0.0f;
#pragma unroll
        for (int r = 0; r < 8; ++r) {
            e0[r] = __expf(s0[r] - mNew);
            e1[r] = __expf(s1[r] - mNew);
            ls += e0[r] + e1[r];
        }
        lRun = lRun * corr + ls + __shfl_xor(ls, 16, 32);

#pragma unroll
        for (int ct = 0; ct < 4; ++ct)
#pragma unroll
            for (int r = 0; r < 8; ++r) acc[ct][r] *= corr;

        // ---- assemble probabilities as bf16 B operand (32x16 layout):
        // vgpr v: lanes<16 -> keys {2v,2v+1}, lanes>=16 -> keys {16+2v,17+2v}.
        // C-layout lane halves are swapped vs B-layout -> one shfl_xor(16).
        v16bf pB;
#pragma unroll
        for (int v = 0; v < 4; ++v) {
            const float o0 = __shfl_xor(e1[2 * v],     16, 32);
            const float o1 = __shfl_xor(e1[2 * v + 1], 16, 32);
            pB[2 * v]     = (__bf16)(hiH ? o0 : e0[2 * v]);
            pB[2 * v + 1] = (__bf16)(hiH ? o1 : e0[2 * v + 1]);
        }
#pragma unroll
        for (int v = 0; v < 4; ++v) {
            const float o0 = __shfl_xor(e0[2 * v],     16, 32);
            const float o1 = __shfl_xor(e0[2 * v + 1], 16, 32);
            pB[8 + 2 * v]     = (__bf16)(hiH ? e1[2 * v]     : o0);
            pB[8 + 2 * v + 1] = (__bf16)(hiH ? e1[2 * v + 1] : o1);
        }

        // ---- out^T += h^T_tile @ P : 4 channel tiles of 16.
        // A 16x32 bf16 layout (contiguous, matching pB): lanes<16 keys 0..15 /
        // lanes>=16 keys 16..31  => two 16B loads.
#pragma unroll
        for (int ct = 0; ct < 4; ++ct) {
            const int ch = ct * 16 + l16;
            const __bf16* hrow = hTb + (size_t)ch * N_ + (size_t)kb * 32 + (hiH ? 16 : 0);
            __builtin_prefetch(hrow + 32, 0, 0);          // next key tile
            const v8bf lo = *(const v8bf*)hrow;           // keys +0..7 (or 16..23)
            const v8bf hi = *(const v8bf*)(hrow + 8);     // keys +8..15 (or 24..31)
            v16bf hA;
#pragma unroll
            for (int i = 0; i < 8; ++i) { hA[i] = lo[i]; hA[8 + i] = hi[i]; }
            acc[ct] = __builtin_amdgcn_wmma_f32_16x16x32_bf16(false, hA, false, pB,
                                                              (short)0, acc[ct],
                                                              false, false);
        }
    }

    // ---- epilogue: out = gamma * (acc/l) + x
    const float gamma = gammaPtr[0];
    const float invL  = 1.0f / lRun;
    const size_t qrow = ((size_t)b * N_ + (size_t)qt * 16 + l16) * C_;
#pragma unroll
    for (int ct = 0; ct < 4; ++ct)
#pragma unroll
        for (int r = 0; r < 8; ++r) {
            const int ch = ct * 16 + r + (hiH ? 8 : 0);
            out[qrow + ch] = fmaf(gamma, acc[ct][r] * invL, x[qrow + ch]);
        }
}

// ---------------------------------------------------------------------------
extern "C" void kernel_launch(void* const* d_in, const int* in_sizes, int n_in,
                              void* d_out, int out_size, void* d_ws, size_t ws_size,
                              hipStream_t stream)
{
    const float* x     = (const float*)d_in[0];
    const float* w_f   = (const float*)d_in[1];
    const float* b_f   = (const float*)d_in[2];
    const float* w_g   = (const float*)d_in[3];
    const float* b_g   = (const float*)d_in[4];
    const float* w_h   = (const float*)d_in[5];
    const float* b_h   = (const float*)d_in[6];
    const float* gamma = (const float*)d_in[7];
    float* out = (float*)d_out;

    char* ws = (char*)d_ws;
    float*  fA = (float*)(ws);                                   // 4*4096*8*4  = 512 KB
    float*  gB = (float*)(ws + (size_t)B_ * N_ * CF_ * 4);       // 512 KB
    __bf16* hT = (__bf16*)(ws + (size_t)2 * B_ * N_ * CF_ * 4);  // 4*64*4096*2 = 2 MB

    proj_kernel<<<B_ * N_, C_, 0, stream>>>(x, w_f, b_f, w_g, b_g, w_h, b_h,
                                            fA, gB, hT);

    // 1024 query tiles (B * N/16), 8 waves per 256-thread block.
    attn_kernel<<<(B_ * (N_ / 16)) / 8, 256, 0, stream>>>(x, fA, gB, hT, gamma, out);
}
